// CompositionAttentionGrammarCell_83966610637459
// MI455X (gfx1250) — compile-verified, matrix operand order checked by reading, not run on hardware
//
#include <hip/hip_runtime.h>

// ---------------------------------------------------------------------------
// CompositionAttentionGrammarCell forward for MI455X (gfx1250, wave32).
// bf16 WMMA (v_wmma_f32_16x16x32_bf16) for every GEMM, fp32 accumulate,
// fp32 VALU + LDS reductions for LayerNorm / softmax attention.
// GEMM: 32x32 register-blocked tile per wave (2 A-frags x 2 B-frags ->
// 4 WMMAs per 32-k step, 2 loads per WMMA), 8 waves/block = 32M x 256N.
// Weight rows prefetched ahead with global_prefetch_b8.
// ---------------------------------------------------------------------------

typedef __bf16 bf16;
typedef __attribute__((ext_vector_type(16))) __bf16 v16bf;
typedef __attribute__((ext_vector_type(8)))  float  v8f;

#define BATCH   32
#define SENT    64
#define DMODEL  768
#define NLAYERS 12
#define NHEADS  12
#define DHEAD   64
#define SS1     129   // SS+1
#define PASTLEN 64
#define SQLEN   65    // SS+1-PAST
#define TSTEPS  10    // MAXC+2
#define NT      30
#define D3      2304  // 3*D
#define D4      3072  // 4*D
#define NROWS   (BATCH*SQLEN)   // 2080 = 65*32, divisible by 32

union FragBF { v16bf v; uint4 u[2]; };

__device__ __forceinline__ float sigmoidf_(float x) { return 1.0f / (1.0f + expf(-x)); }
__device__ __forceinline__ float gelu_tanh_(float v) {
    return 0.5f * v * (1.0f + tanhf(0.7978845608028654f * (v + 0.044715f * v * v * v)));
}

// ---------------------------------------------------------------------------
// Weight conversion kernels (fp32 -> bf16, optionally transposed to N x K)
// ---------------------------------------------------------------------------
__global__ void cvt_bf16_kernel(const float* __restrict__ src, bf16* __restrict__ dst, long n) {
    long i = (long)blockIdx.x * blockDim.x + threadIdx.x;
    if (i < n) dst[i] = (bf16)src[i];
}

// dst[z][c][r] = src[z][r][c]   (src: R x C per batch z)
__global__ void transpose_bf16_kernel(const float* __restrict__ src, bf16* __restrict__ dst,
                                      int R, int C) {
    long i = (long)blockIdx.x * blockDim.x + threadIdx.x;
    long RC = (long)R * C;
    if (i >= RC) return;
    int r = (int)(i / C), c = (int)(i % C);
    long base = (long)blockIdx.z * RC;
    dst[base + (long)c * R + r] = (bf16)src[base + i];
}

// ---------------------------------------------------------------------------
// WMMA GEMM: C[M,N] = A0[M,K0]*Bt0^T + A1[M,K1]*Bt1^T (+bias)(+res)(+act)
// A*: bf16 row-major; Bt*: bf16 N x K row-major (transposed weights).
// One wave -> one 32x32 C tile (acc[mi*2+ni]); block = 8 waves = 256 cols.
// grid = (N/256, M/32), block = 256. act: 0=none 1=relu 2=gelu(tanh)
// ---------------------------------------------------------------------------
__device__ __forceinline__ void wmma_seg(const bf16* __restrict__ A, int lda,
                                         const bf16* __restrict__ Bt, int K,
                                         int mbase, int nbase, int r, int hf,
                                         v8f acc[4]) {
    const bf16* ar0 = A + (long)(mbase + r) * lda;
    const bf16* ar1 = A + (long)(mbase + 16 + r) * lda;
    const bf16* br0 = Bt + (long)(nbase + r) * K;
    const bf16* br1 = Bt + (long)(nbase + 16 + r) * K;
    for (int kb = 0; kb < K; kb += 32) {
        // speculative prefetch ~512B ahead on the streaming weight rows
        __builtin_prefetch((const void*)(br0 + kb + 256), 0, 1);
        __builtin_prefetch((const void*)(br1 + kb + 256), 0, 1);
        FragBF fa0, fa1, fb0, fb1;
        fa0.u[0] = *(const uint4*)(ar0 + kb + 8 * hf);      // k = kb + 8*hf + [0,8)
        fa0.u[1] = *(const uint4*)(ar0 + kb + 16 + 8 * hf); // k = kb + 16 + 8*hf + [0,8)
        fa1.u[0] = *(const uint4*)(ar1 + kb + 8 * hf);
        fa1.u[1] = *(const uint4*)(ar1 + kb + 16 + 8 * hf);
        fb0.u[0] = *(const uint4*)(br0 + kb + 8 * hf);
        fb0.u[1] = *(const uint4*)(br0 + kb + 16 + 8 * hf);
        fb1.u[0] = *(const uint4*)(br1 + kb + 8 * hf);
        fb1.u[1] = *(const uint4*)(br1 + kb + 16 + 8 * hf);
        acc[0] = __builtin_amdgcn_wmma_f32_16x16x32_bf16(false, fa0.v, false, fb0.v,
                                                         (short)0, acc[0], false, false);
        acc[1] = __builtin_amdgcn_wmma_f32_16x16x32_bf16(false, fa0.v, false, fb1.v,
                                                         (short)0, acc[1], false, false);
        acc[2] = __builtin_amdgcn_wmma_f32_16x16x32_bf16(false, fa1.v, false, fb0.v,
                                                         (short)0, acc[2], false, false);
        acc[3] = __builtin_amdgcn_wmma_f32_16x16x32_bf16(false, fa1.v, false, fb1.v,
                                                         (short)0, acc[3], false, false);
    }
}

__global__ void gemm_wmma_bf16(const bf16* __restrict__ A0, int lda0,
                               const bf16* __restrict__ Bt0, int K0,
                               const bf16* __restrict__ A1, int lda1,
                               const bf16* __restrict__ Bt1, int K1,
                               const float* __restrict__ bias,
                               const float* __restrict__ res,
                               float* __restrict__ Cout, bf16* __restrict__ Cbf,
                               int N, int act) {
    int wave  = threadIdx.x >> 5;
    int lane  = threadIdx.x & 31;
    int r  = lane & 15;
    int hf = lane >> 4;
    int nbase = (blockIdx.x * 8 + wave) * 32;
    int mbase = blockIdx.y * 32;

    v8f acc[4] = {{}, {}, {}, {}};
    wmma_seg(A0, lda0, Bt0, K0, mbase, nbase, r, hf, acc);
    if (K1 > 0) wmma_seg(A1, lda1, Bt1, K1, mbase, nbase, r, hf, acc);

#pragma unroll
    for (int mi = 0; mi < 2; ++mi) {
#pragma unroll
        for (int ni = 0; ni < 2; ++ni) {
            int col = nbase + ni * 16 + r;
            float bv = bias ? bias[col] : 0.0f;
#pragma unroll
            for (int i = 0; i < 8; ++i) {
                int row = mbase + mi * 16 + hf * 8 + i;
                float v = acc[mi * 2 + ni][i] + bv;
                if (res)  v += res[(long)row * N + col];
                if (act == 1) v = v > 0.0f ? v : 0.0f;
                else if (act == 2) v = gelu_tanh_(v);
                if (Cout) Cout[(long)row * N + col] = v;
                if (Cbf)  Cbf[(long)row * N + col] = (bf16)v;
            }
        }
    }
}

// ---------------------------------------------------------------------------
// Compose-path kernels
// ---------------------------------------------------------------------------
__global__ void build_elems_kernel(const float* __restrict__ trees,
                                   const int* __restrict__ top_pos,
                                   const int* __restrict__ nt_pos,
                                   bf16* __restrict__ elems_bf,
                                   bf16* __restrict__ rev_bf,
                                   int* __restrict__ lens) {
    int b = blockIdx.x;
    int clen = top_pos[b] - nt_pos[b];
    clen = clen < 1 ? 1 : (clen > 8 ? 8 : clen);
    int len = clen + 2;
    if (threadIdx.x == 0) lens[b] = len;
    const float* tb  = trees + (long)b * SS1 * DMODEL;
    const float* ntv = tb + (long)nt_pos[b] * DMODEL;

    for (int d = threadIdx.x; d < DMODEL; d += blockDim.x) {
        float ev[TSTEPS];
#pragma unroll
        for (int p = 0; p < TSTEPS; ++p) {
            float v;
            if (p == 0) v = ntv[d];
            else if (p <= 8) {
                int ci = nt_pos[b] + p;            // nt_pos + 1 + (p-1)
                ci = ci < 0 ? 0 : (ci > SS1 - 1 ? SS1 - 1 : ci);
                v = tb[(long)ci * DMODEL + d];
            } else v = 0.0f;
            if (p == len - 1) v = ntv[d];
            ev[p] = v;
            elems_bf[((long)b * TSTEPS + p) * DMODEL + d] = (bf16)v;
        }
#pragma unroll
        for (int p = 0; p < TSTEPS; ++p) {
            float v = 0.0f;
            if (p < len) {
                int ri = len - 1 - p;
                ri = ri < 0 ? 0 : (ri > TSTEPS - 1 ? TSTEPS - 1 : ri);
                v = ev[ri];
            }
            rev_bf[((long)b * TSTEPS + p) * DMODEL + d] = (bf16)v;
        }
    }
}

__global__ void zero_state_kernel(float* __restrict__ h, float* __restrict__ c,
                                  bf16* __restrict__ hbf) {
    int i = blockIdx.x * blockDim.x + threadIdx.x;
    if (i < BATCH * DMODEL) { h[i] = 0.0f; c[i] = 0.0f; hbf[i] = (bf16)0.0f; }
}

// gates[b, 0:4D] = [i f g o]; masked LSTM step; write fwd/bwd slot at t==len-2.
__global__ void lstm_point_kernel(const float* __restrict__ gates,
                                  float* __restrict__ h, float* __restrict__ c,
                                  bf16* __restrict__ hbf,
                                  const int* __restrict__ lens, int t,
                                  bf16* __restrict__ outdst /* stride 2D per batch */) {
    int idx = blockIdx.x * blockDim.x + threadIdx.x;
    if (idx >= BATCH * DMODEL) return;
    int b = idx / DMODEL, d = idx % DMODEL;
    const float* g = gates + (long)b * D4;
    float iv = sigmoidf_(g[d]);
    float fv = sigmoidf_(g[DMODEL + d]);
    float gv = tanhf(g[2 * DMODEL + d]);
    float ov = sigmoidf_(g[3 * DMODEL + d]);
    float c2 = fv * c[idx] + iv * gv;
    float h2 = ov * tanhf(c2);
    int len = lens[b];
    if (t == len - 2) outdst[(long)b * (2 * DMODEL) + d] = (bf16)h2;
    if (t < len) { h[idx] = h2; c[idx] = c2; hbf[idx] = (bf16)h2; }
}

// ---------------------------------------------------------------------------
// Build x = updated_trees[:, PAST:, :] + wpe[PAST:SS+1]   (no input mutation)
// ---------------------------------------------------------------------------
__global__ void build_x_kernel(const float* __restrict__ trees,
                               const float* __restrict__ wpe,
                               const float* __restrict__ word_vecs,
                               const float* __restrict__ E_nt,
                               const float* __restrict__ new_child,
                               const int* __restrict__ actions,
                               const int* __restrict__ pointer,
                               const int* __restrict__ top_pos,
                               const int* __restrict__ nt_pos,
                               float* __restrict__ x) {
    int row = blockIdx.x;                 // 0..NROWS-1
    int b = row / SQLEN, q = row % SQLEN;
    int pos = PASTLEN + q;
    int a = actions[b];
    bool is_shift = (a == 0);
    bool push = is_shift || (a >= 2);
    int wp = push ? top_pos[b] + 1 : nt_pos[b];
    int nti = a - 2; nti = nti < 0 ? 0 : (nti > NT - 1 ? NT - 1 : nti);
    for (int d = threadIdx.x; d < DMODEL; d += blockDim.x) {
        float v;
        if (pos == wp) {
            if (push)
                v = is_shift ? word_vecs[((long)b * SENT + pointer[b]) * DMODEL + d]
                             : E_nt[(long)nti * DMODEL + d];
            else
                v = new_child[(long)b * DMODEL + d];
        } else {
            v = trees[((long)b * SS1 + pos) * DMODEL + d];
        }
        x[(long)row * DMODEL + d] = v + wpe[(long)pos * DMODEL + d];
    }
}

// ---------------------------------------------------------------------------
// LayerNorm -> bf16  (one block per row)
// ---------------------------------------------------------------------------
__global__ void ln_bf16_kernel(const float* __restrict__ x,
                               const float* __restrict__ gamma,
                               const float* __restrict__ beta,
                               bf16* __restrict__ out) {
    __shared__ float red[256];
    int row = blockIdx.x, tid = threadIdx.x;
    const float* xr = x + (long)row * DMODEL;
    float s = 0.0f;
    for (int d = tid; d < DMODEL; d += 256) s += xr[d];
    red[tid] = s; __syncthreads();
    for (int o = 128; o > 0; o >>= 1) { if (tid < o) red[tid] += red[tid + o]; __syncthreads(); }
    float mean = red[0] * (1.0f / DMODEL);
    __syncthreads();
    float vs = 0.0f;
    for (int d = tid; d < DMODEL; d += 256) { float t = xr[d] - mean; vs += t * t; }
    red[tid] = vs; __syncthreads();
    for (int o = 128; o > 0; o >>= 1) { if (tid < o) red[tid] += red[tid + o]; __syncthreads(); }
    float inv = rsqrtf(red[0] * (1.0f / DMODEL) + 1e-5f);
    for (int d = tid; d < DMODEL; d += 256)
        out[(long)row * DMODEL + d] = (bf16)((xr[d] - mean) * inv * gamma[d] + beta[d]);
}

// ---------------------------------------------------------------------------
// Attention: fp32 scores/softmax, bf16 ctx output. grid=(SQ, NH, B), block=256.
// ---------------------------------------------------------------------------
__global__ void attn_kernel(const float* __restrict__ qkv,
                            const float* __restrict__ keys,
                            const float* __restrict__ values,
                            bf16* __restrict__ ctx_bf, int l) {
    __shared__ float qs[DHEAD];
    __shared__ float sc[SS1];
    __shared__ float red[256];
    int q = blockIdx.x, h = blockIdx.y, b = blockIdx.z;
    int row = b * SQLEN + q;
    int tid = threadIdx.x;

    if (tid < DHEAD) qs[tid] = qkv[(long)row * D3 + h * DHEAD + tid];
    __syncthreads();

    float myscore = -3.4e38f;
    if (tid < SS1) {
        float s;
        if (tid <= PASTLEN + q) {
            const float* kp;
            if (tid < PASTLEN)
                kp = keys + ((((long)b * SS1 + tid) * NLAYERS + l) * NHEADS + h) * DHEAD;
            else
                kp = qkv + (long)(b * SQLEN + (tid - PASTLEN)) * D3 + DMODEL + h * DHEAD;
            float dot = 0.0f;
            for (int d = 0; d < DHEAD; ++d) dot += qs[d] * kp[d];
            s = dot * 0.125f;   // 1/sqrt(64)
        } else {
            s = -1.0e9f;
        }
        sc[tid] = s;
        myscore = s;
    }
    red[tid] = myscore; __syncthreads();
    for (int o = 128; o > 0; o >>= 1) { if (tid < o) red[tid] = fmaxf(red[tid], red[tid + o]); __syncthreads(); }
    float m = red[0];
    __syncthreads();
    float e = 0.0f;
    if (tid < SS1) { e = expf(sc[tid] - m); sc[tid] = e; }
    red[tid] = e; __syncthreads();
    for (int o = 128; o > 0; o >>= 1) { if (tid < o) red[tid] += red[tid + o]; __syncthreads(); }
    float inv = 1.0f / red[0];
    __syncthreads();

    if (tid < DHEAD) {
        float acc = 0.0f;
        for (int t = 0; t < SS1; ++t) {
            const float* vp;
            if (t < PASTLEN)
                vp = values + ((((long)b * SS1 + t) * NLAYERS + l) * NHEADS + h) * DHEAD;
            else
                vp = qkv + (long)(b * SQLEN + (t - PASTLEN)) * D3 + 2 * DMODEL + h * DHEAD;
            acc += sc[t] * vp[tid];
        }
        ctx_bf[(long)row * DMODEL + h * DHEAD + tid] = (bf16)(acc * inv);
    }
}

// ---------------------------------------------------------------------------
// Final LN + gather at target row -> d_out (fp32)
// ---------------------------------------------------------------------------
__global__ void final_kernel(const float* __restrict__ x,
                             const float* __restrict__ gamma,
                             const float* __restrict__ beta,
                             const int* __restrict__ actions,
                             const int* __restrict__ top_pos,
                             const int* __restrict__ nt_pos,
                             float* __restrict__ out) {
    __shared__ float red[256];
    int b = blockIdx.x, tid = threadIdx.x;
    int a = actions[b];
    bool push = (a == 0) || (a >= 2);
    int wp = push ? top_pos[b] + 1 : nt_pos[b];
    int row = b * SQLEN + (wp - PASTLEN);
    const float* xr = x + (long)row * DMODEL;
    float s = 0.0f;
    for (int d = tid; d < DMODEL; d += 256) s += xr[d];
    red[tid] = s; __syncthreads();
    for (int o = 128; o > 0; o >>= 1) { if (tid < o) red[tid] += red[tid + o]; __syncthreads(); }
    float mean = red[0] * (1.0f / DMODEL);
    __syncthreads();
    float vs = 0.0f;
    for (int d = tid; d < DMODEL; d += 256) { float t = xr[d] - mean; vs += t * t; }
    red[tid] = vs; __syncthreads();
    for (int o = 128; o > 0; o >>= 1) { if (tid < o) red[tid] += red[tid + o]; __syncthreads(); }
    float inv = rsqrtf(red[0] * (1.0f / DMODEL) + 1e-5f);
    for (int d = tid; d < DMODEL; d += 256)
        out[(long)b * DMODEL + d] = (xr[d] - mean) * inv * gamma[d] + beta[d];
}

// ---------------------------------------------------------------------------
// Host launcher
// ---------------------------------------------------------------------------
static inline void* bump(char*& p, size_t bytes) {
    void* r = (void*)p;
    p += (bytes + 255) & ~(size_t)255;
    return r;
}

extern "C" void kernel_launch(void* const* d_in, const int* in_sizes, int n_in,
                              void* d_out, int out_size, void* d_ws, size_t ws_size,
                              hipStream_t stream) {
    const float* word_vecs = (const float*)d_in[0];
    const float* trees     = (const float*)d_in[1];
    const float* keys      = (const float*)d_in[2];
    const float* values    = (const float*)d_in[3];
    const float* E_nt      = (const float*)d_in[4];
    const float* wpe       = (const float*)d_in[5];
    const float* ln1_g     = (const float*)d_in[6];
    const float* ln1_b     = (const float*)d_in[7];
    const float* attn_w    = (const float*)d_in[8];
    const float* attn_b    = (const float*)d_in[9];
    const float* proj_w    = (const float*)d_in[10];
    const float* proj_b    = (const float*)d_in[11];
    const float* ln2_g     = (const float*)d_in[12];
    const float* ln2_b     = (const float*)d_in[13];
    const float* fc_w      = (const float*)d_in[14];
    const float* fc_b      = (const float*)d_in[15];
    const float* fc2_w     = (const float*)d_in[16];
    const float* fc2_b     = (const float*)d_in[17];
    const float* lnf_g     = (const float*)d_in[18];
    const float* lnf_b     = (const float*)d_in[19];
    const float* Wih_f     = (const float*)d_in[20];
    const float* Whh_f     = (const float*)d_in[21];
    const float* b_f       = (const float*)d_in[22];
    const float* Wih_b     = (const float*)d_in[23];
    const float* Whh_b     = (const float*)d_in[24];
    const float* b_b       = (const float*)d_in[25];
    const float* comp_w    = (const float*)d_in[26];
    const float* comp_b    = (const float*)d_in[27];
    const int*   actions   = (const int*)d_in[28];
    const int*   pointer   = (const int*)d_in[29];
    const int*   top_pos   = (const int*)d_in[30];
    const int*   nt_pos    = (const int*)d_in[31];
    float* out = (float*)d_out;

    // ---- workspace layout (~236 MB) ----
    char* p = (char*)d_ws;
    bf16* attn_wT = (bf16*)bump(p, (size_t)NLAYERS * D3 * DMODEL * 2);
    bf16* proj_wT = (bf16*)bump(p, (size_t)NLAYERS * DMODEL * DMODEL * 2);
    bf16* fc_wT   = (bf16*)bump(p, (size_t)NLAYERS * D4 * DMODEL * 2);
    bf16* fc2_wT  = (bf16*)bump(p, (size_t)NLAYERS * DMODEL * D4 * 2);
    bf16* wih_f   = (bf16*)bump(p, (size_t)D4 * DMODEL * 2);
    bf16* whh_f   = (bf16*)bump(p, (size_t)D4 * DMODEL * 2);
    bf16* wih_b   = (bf16*)bump(p, (size_t)D4 * DMODEL * 2);
    bf16* whh_b   = (bf16*)bump(p, (size_t)D4 * DMODEL * 2);
    bf16* comp_wT = (bf16*)bump(p, (size_t)DMODEL * 2 * DMODEL * 2);
    bf16* elems_bf = (bf16*)bump(p, (size_t)BATCH * TSTEPS * DMODEL * 2);
    bf16* rev_bf   = (bf16*)bump(p, (size_t)BATCH * TSTEPS * DMODEL * 2);
    int*  lens     = (int*)bump(p, BATCH * sizeof(int));
    float* hstate  = (float*)bump(p, (size_t)BATCH * DMODEL * 4);
    float* cstate  = (float*)bump(p, (size_t)BATCH * DMODEL * 4);
    bf16*  hbf     = (bf16*)bump(p, (size_t)BATCH * DMODEL * 2);
    float* gates   = (float*)bump(p, (size_t)BATCH * D4 * 4);
    bf16*  fwdbwd  = (bf16*)bump(p, (size_t)BATCH * 2 * DMODEL * 2);
    float* newchild = (float*)bump(p, (size_t)BATCH * DMODEL * 4);
    float* x        = (float*)bump(p, (size_t)NROWS * DMODEL * 4);
    bf16*  abf      = (bf16*)bump(p, (size_t)NROWS * DMODEL * 2);
    float* qkv      = (float*)bump(p, (size_t)NROWS * D3 * 4);
    bf16*  ctxbf    = (bf16*)bump(p, (size_t)NROWS * DMODEL * 2);
    bf16*  mlpbf    = (bf16*)bump(p, (size_t)NROWS * D4 * 2);
    (void)ws_size; (void)n_in; (void)in_sizes; (void)out_size;

    // ---- 1) weight conversion (fp32 -> bf16, N x K) ----
    {
        long n = (long)DMODEL * D3;  // per layer
        transpose_bf16_kernel<<<dim3((n + 255) / 256, 1, NLAYERS), 256, 0, stream>>>(attn_w, attn_wT, DMODEL, D3);
        n = (long)DMODEL * DMODEL;
        transpose_bf16_kernel<<<dim3((n + 255) / 256, 1, NLAYERS), 256, 0, stream>>>(proj_w, proj_wT, DMODEL, DMODEL);
        n = (long)DMODEL * D4;
        transpose_bf16_kernel<<<dim3((n + 255) / 256, 1, NLAYERS), 256, 0, stream>>>(fc_w, fc_wT, DMODEL, D4);
        transpose_bf16_kernel<<<dim3((n + 255) / 256, 1, NLAYERS), 256, 0, stream>>>(fc2_w, fc2_wT, D4, DMODEL);
        long m = (long)D4 * DMODEL;
        cvt_bf16_kernel<<<(m + 255) / 256, 256, 0, stream>>>(Wih_f, wih_f, m);
        cvt_bf16_kernel<<<(m + 255) / 256, 256, 0, stream>>>(Whh_f, whh_f, m);
        cvt_bf16_kernel<<<(m + 255) / 256, 256, 0, stream>>>(Wih_b, wih_b, m);
        cvt_bf16_kernel<<<(m + 255) / 256, 256, 0, stream>>>(Whh_b, whh_b, m);
        long cw = (long)2 * DMODEL * DMODEL;
        transpose_bf16_kernel<<<dim3((cw + 255) / 256, 1, 1), 256, 0, stream>>>(comp_w, comp_wT, 2 * DMODEL, DMODEL);
    }

    // ---- 2) compose: build elems/rev, run fwd + bwd LSTM with WMMA gates ----
    build_elems_kernel<<<BATCH, 256, 0, stream>>>(trees, top_pos, nt_pos, elems_bf, rev_bf, lens);

    const int npt = (BATCH * DMODEL + 255) / 256;
    zero_state_kernel<<<npt, 256, 0, stream>>>(hstate, cstate, hbf);
    for (int t = 0; t < TSTEPS; ++t) {
        gemm_wmma_bf16<<<dim3(D4 / 256, BATCH / 32), 256, 0, stream>>>(
            elems_bf + (long)t * DMODEL, TSTEPS * DMODEL, wih_f, DMODEL,
            hbf, DMODEL, whh_f, DMODEL,
            b_f, nullptr, gates, nullptr, D4, 0);
        lstm_point_kernel<<<npt, 256, 0, stream>>>(gates, hstate, cstate, hbf, lens, t, fwdbwd);
    }
    zero_state_kernel<<<npt, 256, 0, stream>>>(hstate, cstate, hbf);
    for (int t = 0; t < TSTEPS; ++t) {
        gemm_wmma_bf16<<<dim3(D4 / 256, BATCH / 32), 256, 0, stream>>>(
            rev_bf + (long)t * DMODEL, TSTEPS * DMODEL, wih_b, DMODEL,
            hbf, DMODEL, whh_b, DMODEL,
            b_b, nullptr, gates, nullptr, D4, 0);
        lstm_point_kernel<<<npt, 256, 0, stream>>>(gates, hstate, cstate, hbf, lens, t, fwdbwd + DMODEL);
    }
    // new_child = relu([fwd|bwd] @ comp_w + comp_b)
    gemm_wmma_bf16<<<dim3(DMODEL / 256, BATCH / 32), 256, 0, stream>>>(
        fwdbwd, 2 * DMODEL, comp_wT, 2 * DMODEL,
        nullptr, 0, nullptr, 0,
        comp_b, nullptr, newchild, nullptr, DMODEL, 1);

    // ---- 3) stack update + positional embedding -> x ----
    build_x_kernel<<<NROWS, 256, 0, stream>>>(trees, wpe, word_vecs, E_nt, newchild,
                                              actions, pointer, top_pos, nt_pos, x);

    // ---- 4) transformer layers ----
    for (int l = 0; l < NLAYERS; ++l) {
        ln_bf16_kernel<<<NROWS, 256, 0, stream>>>(x, ln1_g + (long)l * DMODEL, ln1_b + (long)l * DMODEL, abf);
        gemm_wmma_bf16<<<dim3(D3 / 256, NROWS / 32), 256, 0, stream>>>(
            abf, DMODEL, attn_wT + (long)l * D3 * DMODEL, DMODEL,
            nullptr, 0, nullptr, 0,
            attn_b + (long)l * D3, nullptr, qkv, nullptr, D3, 0);
        attn_kernel<<<dim3(SQLEN, NHEADS, BATCH), 256, 0, stream>>>(qkv, keys, values, ctxbf, l);
        gemm_wmma_bf16<<<dim3(DMODEL / 256, NROWS / 32), 256, 0, stream>>>(
            ctxbf, DMODEL, proj_wT + (long)l * DMODEL * DMODEL, DMODEL,
            nullptr, 0, nullptr, 0,
            proj_b + (long)l * DMODEL, x, x, nullptr, DMODEL, 0);
        ln_bf16_kernel<<<NROWS, 256, 0, stream>>>(x, ln2_g + (long)l * DMODEL, ln2_b + (long)l * DMODEL, abf);
        gemm_wmma_bf16<<<dim3(D4 / 256, NROWS / 32), 256, 0, stream>>>(
            abf, DMODEL, fc_wT + (long)l * D4 * DMODEL, DMODEL,
            nullptr, 0, nullptr, 0,
            fc_b + (long)l * D4, nullptr, nullptr, mlpbf, D4, 2);
        gemm_wmma_bf16<<<dim3(DMODEL / 256, NROWS / 32), 256, 0, stream>>>(
            mlpbf, D4, fc2_wT + (long)l * DMODEL * D4, D4,
            nullptr, 0, nullptr, 0,
            fc2_b + (long)l * DMODEL, x, x, nullptr, DMODEL, 0);
    }

    // ---- 5) final LN + gather ----
    final_kernel<<<BATCH, 256, 0, stream>>>(x, lnf_g, lnf_b, actions, top_pos, nt_pos, out);
}
